// MultiHeadAttention_8667244003725
// MI455X (gfx1250) — matrix operand order repaired and checked by reading, not compiled, not run on hardware
//
#include <hip/hip_runtime.h>
#include <stdint.h>

// Problem constants (match reference)
#define B_ 4
#define S_ 1024
#define E_ 1024
#define H_ 16
#define D_ 64

typedef __bf16 bf16;
typedef __attribute__((ext_vector_type(16))) __bf16 bf16x16;
typedef __attribute__((ext_vector_type(8)))  __bf16 bf16x8;
typedef __attribute__((ext_vector_type(8)))  float   f32x8;

union FragHalves { bf16x16 v; bf16x8 h[2]; };

// ---------------------------------------------------------------------------
// WMMA helpers (CDNA5 wave32, V_WMMA_F32_16X16X32_BF16)
// A fragment: 16x32 (MxK) bf16 from row-major source (K contiguous).
//   lane m (0-15): row m, K = 0..7 in h[0], K = 16..23 in h[1]
//   lane m+16    : row m, K = 8..15 in h[0], K = 24..31 in h[1]
__device__ __forceinline__ bf16x16 load_frag_a(const bf16* base, int stride) {
  int lane = threadIdx.x & 31;
  int m = lane & 15;
  int kb = (lane >> 4) ? 8 : 0;
  const bf16* p = base + m * stride + kb;
  FragHalves f;
  f.h[0] = *reinterpret_cast<const bf16x8*>(p);
  f.h[1] = *reinterpret_cast<const bf16x8*>(p + 16);
  return f.v;
}

// B fragment: 32x16 (KxN) bf16 from N-major source srcT[n][k] (K contiguous).
//   lane n (0-15): col n, K = 0..7 in h[0], K = 16..23 in h[1]; lane n+16: col n, K = 8..15 in h[0], K = 24..31 in h[1]
__device__ __forceinline__ bf16x16 load_frag_b(const bf16* base, int stride) {
  int lane = threadIdx.x & 31;
  int n = lane & 15;
  int kb = (lane >> 4) ? 8 : 0;
  const bf16* p = base + n * stride + kb;
  FragHalves f;
  f.h[0] = *reinterpret_cast<const bf16x8*>(p);
  f.h[1] = *reinterpret_cast<const bf16x8*>(p + 16);
  return f.v;
}

__device__ __forceinline__ f32x8 wmma_bf16(bf16x16 a, bf16x16 b, f32x8 c) {
  return __builtin_amdgcn_wmma_f32_16x16x32_bf16(false, a, false, b,
                                                 (short)0, c, false, false);
}

// Async cache->LDS copy (CDNA5 GLOBAL_LOAD_ASYNC_TO_LDS_B128, ASYNCcnt).
// ldsoff: per-lane LDS byte address (low 32 bits of a generic LDS pointer),
// gptr: per-lane 64-bit global address.
__device__ __forceinline__ void async_load_b128(uint32_t ldsoff,
                                                const void* gptr) {
  asm volatile("global_load_async_to_lds_b128 %0, %1, off"
               :: "v"(ldsoff), "v"(gptr) : "memory");
}
__device__ __forceinline__ void wait_async0() {
  asm volatile("s_wait_asynccnt 0" ::: "memory");
}

// ---------------------------------------------------------------------------
// Conversion kernels (streaming: non-temporal reads)
__global__ void cvt_f32_to_bf16(const float* __restrict__ in,
                                bf16* __restrict__ out, int n) {
  int i = blockIdx.x * 256 + threadIdx.x;
  if (i < n) out[i] = (bf16)__builtin_nontemporal_load(&in[i]);
}

// W[h][e][d] (f32) -> Wt[h*D+d][e] (bf16), N-major for GEMM B operand
__global__ void cvt_w_qkv(const float* __restrict__ W, bf16* __restrict__ out) {
  size_t i = (size_t)blockIdx.x * 256 + threadIdx.x;  // over H*D*E
  int e = (int)(i % E_);
  size_t n = i / E_;
  int d = (int)(n % D_);
  int h = (int)(n / D_);
  out[i] = (bf16)__builtin_nontemporal_load(&W[((size_t)h * E_ + e) * D_ + d]);
}

// bf16 transpose per batch: in[b][r][c] -> out[b][c][r]
__global__ __launch_bounds__(256)
void transpose_bf16(const bf16* __restrict__ in, bf16* __restrict__ out,
                    int rows, int cols) {
  __shared__ bf16 t[32][33];
  int bx = blockIdx.x * 32, by = blockIdx.y * 32;
  const bf16* ib = in + (size_t)blockIdx.z * rows * cols;
  bf16* ob = out + (size_t)blockIdx.z * rows * cols;
  int x = threadIdx.x;
  for (int y = threadIdx.y; y < 32; y += 8)
    t[y][x] = ib[(size_t)(by + y) * cols + (bx + x)];
  __syncthreads();
  for (int y = threadIdx.y; y < 32; y += 8)
    ob[(size_t)(bx + y) * rows + (by + x)] = t[x][y];
}

// ---------------------------------------------------------------------------
// Tiled bf16 GEMM: C[M x N] = A[M x K] (row-major) * Bt[N x K] (N-major)
// Block tile 64x64x32, 4 waves (2x2), each wave 32x32 = 2x2 WMMA frags.
// A/B tiles staged to LDS with async cache->LDS copies (no VGPR round-trip).
template <bool OUTF32>
__global__ __launch_bounds__(128)
void gemm_bt(const bf16* __restrict__ A, const bf16* __restrict__ Bt,
             void* __restrict__ Cv, int M, int N, int K,
             long long sA, long long sB, long long sC) {
  __shared__ bf16 As[64][48];  // stride 48 elems = 96B (16B-aligned rows)
  __shared__ bf16 Bs[64][48];
  const bf16* Ab = A + (long long)blockIdx.z * sA;
  const bf16* Bb = Bt + (long long)blockIdx.z * sB;
  int m0 = blockIdx.y * 64, n0 = blockIdx.x * 64;
  int tid = threadIdx.x;
  int wid = tid >> 5;
  int wm = (wid >> 1) * 32, wn = (wid & 1) * 32;
  int r = tid >> 1, hf = (tid & 1) * 16;

  uint32_t ldsA = (uint32_t)(uintptr_t)&As[r][hf];
  uint32_t ldsB = (uint32_t)(uintptr_t)&Bs[r][hf];

  f32x8 acc[2][2] = {};

  for (int kk = 0; kk < K; kk += 32) {
    const bf16* ga = Ab + (size_t)(m0 + r) * K + kk + hf;
    const bf16* gb = Bb + (size_t)(n0 + r) * K + kk + hf;
    async_load_b128(ldsA, ga);
    async_load_b128(ldsA + 16, ga + 8);
    async_load_b128(ldsB, gb);
    async_load_b128(ldsB + 16, gb + 8);
    if (kk + 32 < K) {  // prefetch next k-tile -> global_prefetch_b8
      __builtin_prefetch(ga + 32, 0, 1);
      __builtin_prefetch(gb + 32, 0, 1);
    }
    wait_async0();
    __syncthreads();

    bf16x16 af0 = load_frag_a(&As[wm][0], 48);
    bf16x16 af1 = load_frag_a(&As[wm + 16][0], 48);
    bf16x16 bf0 = load_frag_b(&Bs[wn][0], 48);
    bf16x16 bf1 = load_frag_b(&Bs[wn + 16][0], 48);
    acc[0][0] = wmma_bf16(af0, bf0, acc[0][0]);
    acc[0][1] = wmma_bf16(af0, bf1, acc[0][1]);
    acc[1][0] = wmma_bf16(af1, bf0, acc[1][0]);
    acc[1][1] = wmma_bf16(af1, bf1, acc[1][1]);
    __syncthreads();  // waits DS reads, protects WAR vs next async writes
  }

  int lane = tid & 31;
  int lc = lane & 15;
  int hi8 = (lane >> 4) * 8;
#pragma unroll
  for (int i = 0; i < 2; ++i)
#pragma unroll
    for (int j = 0; j < 2; ++j)
#pragma unroll
      for (int rr = 0; rr < 8; ++rr) {
        int row = m0 + wm + i * 16 + rr + hi8;
        int col = n0 + wn + j * 16 + lc;
        if constexpr (OUTF32) {
          float* Cb = (float*)Cv + (long long)blockIdx.z * sC;
          Cb[(size_t)row * N + col] = acc[i][j][rr];
        } else {
          bf16* Cb = (bf16*)Cv + (long long)blockIdx.z * sC;
          Cb[(size_t)row * N + col] = (bf16)acc[i][j][rr];
        }
      }
}

// ---------------------------------------------------------------------------
// Flash attention: one wave handles one (b, h, 16-query block).
// Q[b][s][h*64+d], K[b][s][h*64+d] (bf16, row-major over d),
// Vt[b][h*64+d][s] (bf16, N-major for P*V), output Ct[b][h*64+d][s] (bf16,
// already transposed for the final W_O @ C GEMM).
__global__ __launch_bounds__(256)
void attn_kernel(const bf16* __restrict__ Q, const bf16* __restrict__ Km,
                 const bf16* __restrict__ Vt, bf16* __restrict__ Ct) {
  __shared__ bf16 plds[8][16 * 32];  // wave-private P re-layout buffers
  int wid = threadIdx.x >> 5;
  int gw = blockIdx.x * 8 + wid;
  int qb = gw & 63;          // 64 query blocks of 16
  int h = (gw >> 6) & 15;
  int b = gw >> 10;
  int lane = threadIdx.x & 31;
  int lc = lane & 15;
  int hi8 = (lane >> 4) * 8;

  const bf16* Qb = Q + ((size_t)b * S_) * E_ + h * D_;
  const bf16* Kb = Km + ((size_t)b * S_) * E_ + h * D_;
  const bf16* Vb = Vt + ((size_t)b * E_ + h * D_) * S_;
  bf16* Ob = Ct + ((size_t)b * E_ + h * D_) * S_;

  int q0 = qb * 16;
  bf16x16 aq0 = load_frag_a(Qb + (size_t)q0 * E_, E_);       // d = 0..31
  bf16x16 aq1 = load_frag_a(Qb + (size_t)q0 * E_ + 32, E_);  // d = 32..63

  f32x8 o[4] = {};
  float mrow[8], lrow[8];
#pragma unroll
  for (int rr = 0; rr < 8; ++rr) { mrow[rr] = -1e30f; lrow[rr] = 0.f; }

  bf16* pl = &plds[wid][0];
  int nkeys = q0 + 16;              // causal: keys 0 .. q0+15
  int nch = (nkeys + 31) >> 5;      // 32-key chunks

  for (int ch = 0; ch < nch; ++ch) {
    int kk0 = ch << 5;
    f32x8 sv[2];
#pragma unroll
    for (int sub = 0; sub < 2; ++sub) {
      int k0 = kk0 + sub * 16;
      f32x8 s = {};
      if (k0 < nkeys) {
        bf16x16 bk0 = load_frag_b(Kb + (size_t)k0 * E_, E_);       // d 0..31
        bf16x16 bk1 = load_frag_b(Kb + (size_t)k0 * E_ + 32, E_);  // d 32..63
        s = wmma_bf16(aq0, bk0, s);
        s = wmma_bf16(aq1, bk1, s);
#pragma unroll
        for (int rr = 0; rr < 8; ++rr) {
          int qrow = q0 + rr + hi8;
          int kcol = k0 + lc;
          s[rr] = (qrow >= kcol) ? s[rr] * 0.125f : -1e30f;  // 1/sqrt(64)
        }
      } else {
#pragma unroll
        for (int rr = 0; rr < 8; ++rr) s[rr] = -1e30f;
      }
      sv[sub] = s;
    }

    // online softmax: row max across 16 lanes of each half-wave
    float mnew[8], alpha[8];
#pragma unroll
    for (int rr = 0; rr < 8; ++rr) {
      float mx = fmaxf(sv[0][rr], sv[1][rr]);
      mx = fmaxf(mx, __shfl_xor(mx, 1, 32));
      mx = fmaxf(mx, __shfl_xor(mx, 2, 32));
      mx = fmaxf(mx, __shfl_xor(mx, 4, 32));
      mx = fmaxf(mx, __shfl_xor(mx, 8, 32));
      mnew[rr] = fmaxf(mrow[rr], mx);
      alpha[rr] = __expf(mrow[rr] - mnew[rr]);
      mrow[rr] = mnew[rr];
    }

    // P = exp(S - m); stash into LDS (C-layout -> A-layout transpose)
    float psum[8];
#pragma unroll
    for (int rr = 0; rr < 8; ++rr) psum[rr] = 0.f;
#pragma unroll
    for (int sub = 0; sub < 2; ++sub)
#pragma unroll
      for (int rr = 0; rr < 8; ++rr) {
        float p = __expf(sv[sub][rr] - mnew[rr]);
        psum[rr] += p;
        pl[(rr + hi8) * 32 + sub * 16 + lc] = (bf16)p;
      }
#pragma unroll
    for (int rr = 0; rr < 8; ++rr) {
      float t = psum[rr];
      t += __shfl_xor(t, 1, 32);
      t += __shfl_xor(t, 2, 32);
      t += __shfl_xor(t, 4, 32);
      t += __shfl_xor(t, 8, 32);
      lrow[rr] = lrow[rr] * alpha[rr] + t;
    }
#pragma unroll
    for (int j = 0; j < 4; ++j)
#pragma unroll
      for (int rr = 0; rr < 8; ++rr) o[j][rr] *= alpha[rr];

    asm volatile("s_wait_dscnt 0" ::: "memory");  // P stores -> A-frag loads
    bf16x16 ap = load_frag_a(pl, 32);             // P: 16 x 32 keys
#pragma unroll
    for (int j = 0; j < 4; ++j) {                 // d tiles of 16
      bf16x16 bv = load_frag_b(Vb + (size_t)(j * 16) * S_ + kk0, S_);
      o[j] = wmma_bf16(ap, bv, o[j]);
    }
    asm volatile("s_wait_dscnt 0" ::: "memory");  // guard LDS WAR next iter
  }

  // normalize and store transposed: Ct[b][h*64+d][s]
#pragma unroll
  for (int j = 0; j < 4; ++j)
#pragma unroll
    for (int rr = 0; rr < 8; ++rr) {
      float v = o[j][rr] / lrow[rr];
      int srow = q0 + rr + hi8;
      int dcol = j * 16 + lc;
      Ob[(size_t)dcol * S_ + srow] = (bf16)v;
    }
}

// ---------------------------------------------------------------------------
extern "C" void kernel_launch(void* const* d_in, const int* in_sizes, int n_in,
                              void* d_out, int out_size, void* d_ws,
                              size_t ws_size, hipStream_t stream) {
  (void)in_sizes; (void)n_in; (void)out_size; (void)ws_size;
  const float* x  = (const float*)d_in[0];
  const float* Wq = (const float*)d_in[1];
  const float* Wk = (const float*)d_in[2];
  const float* Wv = (const float*)d_in[3];
  const float* Wo = (const float*)d_in[4];

  char* ws = (char*)d_ws;
  size_t off = 0;
  auto take = [&](size_t bytes) {
    char* p = ws + off;
    off += (bytes + 255) & ~(size_t)255;
    return p;
  };
  bf16* xb  = (bf16*)take((size_t)B_ * S_ * E_ * 2);
  bf16* wqt = (bf16*)take((size_t)E_ * E_ * 2);
  bf16* wkt = (bf16*)take((size_t)E_ * E_ * 2);
  bf16* wvt = (bf16*)take((size_t)E_ * E_ * 2);
  bf16* wob = (bf16*)take((size_t)E_ * E_ * 2);
  bf16* Qb  = (bf16*)take((size_t)B_ * S_ * E_ * 2);
  bf16* Kb  = (bf16*)take((size_t)B_ * S_ * E_ * 2);
  bf16* Vb  = (bf16*)take((size_t)B_ * S_ * E_ * 2);
  bf16* Vtb = (bf16*)take((size_t)B_ * S_ * E_ * 2);
  bf16* Ctb = (bf16*)take((size_t)B_ * S_ * E_ * 2);

  // 1) conversions
  {
    int n = B_ * S_ * E_;
    cvt_f32_to_bf16<<<n / 256, 256, 0, stream>>>(x, xb, n);
  }
  {
    int n = E_ * E_;
    cvt_w_qkv<<<n / 256, 256, 0, stream>>>(Wq, wqt);
    cvt_w_qkv<<<n / 256, 256, 0, stream>>>(Wk, wkt);
    cvt_w_qkv<<<n / 256, 256, 0, stream>>>(Wv, wvt);
    cvt_f32_to_bf16<<<n / 256, 256, 0, stream>>>(Wo, wob, n);
  }

  // 2) QKV projections: per batch, [S x E] = [S x E] * Wt[E x E]^T
  dim3 gp(E_ / 64, S_ / 64, B_);
  long long se = (long long)S_ * E_;
  gemm_bt<false><<<gp, 128, 0, stream>>>(xb, wqt, Qb, S_, E_, E_, se, 0, se);
  gemm_bt<false><<<gp, 128, 0, stream>>>(xb, wkt, Kb, S_, E_, E_, se, 0, se);
  gemm_bt<false><<<gp, 128, 0, stream>>>(xb, wvt, Vb, S_, E_, E_, se, 0, se);

  // 3) V transpose: Vt[b][e][s] = V[b][s][e]
  transpose_bf16<<<dim3(E_ / 32, S_ / 32, B_), dim3(32, 8), 0, stream>>>(
      Vb, Vtb, S_, E_);

  // 4) flash attention (writes Ct[b][e][s], i.e. concat^T per batch)
  attn_kernel<<<(B_ * H_ * (S_ / 16)) / 8, 256, 0, stream>>>(Qb, Kb, Vtb, Ctb);

  // 5) output projection: out[b] = W_O @ C[b]  (f32 output)
  gemm_bt<true><<<dim3(E_ / 64, E_ / 64, B_), 128, 0, stream>>>(
      wob, Ctb, d_out, E_, E_, S_, 0, (long long)E_ * S_,
      (long long)E_ * E_);
}
